// MultiHeadAttentionBlock_51402168598763
// MI455X (gfx1250) — matrix operand order repaired and checked by reading, not compiled, not run on hardware
//
#include <hip/hip_runtime.h>
#include <hip/hip_bf16.h>

// ---------------------------------------------------------------------------
// MHA block for MI455X (gfx1250, wave32, WMMA bf16 16x16x32 w/ f32 accum).
// Pipeline: cast->bf16, QKV GEMMs with TDM (tensor_load_to_lds) double-buffered
// LDS staging (V stored head-transposed), flash attention with permlane16
// softmax reductions, output projection GEMM (fp32 out).
// ---------------------------------------------------------------------------

#define B_  4
#define S_  2048
#define D_  512
#define H_  8
#define DK_ 64

typedef __attribute__((ext_vector_type(16))) __bf16 v16bf;
typedef __attribute__((ext_vector_type(8)))  __bf16 v8bf;
typedef __attribute__((ext_vector_type(4)))  __bf16 v4bf;
typedef __attribute__((ext_vector_type(8)))  float  v8f;
typedef __attribute__((ext_vector_type(4)))  unsigned int v4u;
typedef __attribute__((ext_vector_type(8)))  int    v8i;
typedef __attribute__((ext_vector_type(4)))  int    v4i;

union V16 { v16bf v; v8bf h[2]; };

#if __has_builtin(__builtin_amdgcn_tensor_load_to_lds) && __has_builtin(__builtin_amdgcn_s_wait_tensorcnt)
#define USE_TDM 1
#else
#define USE_TDM 0
#warning "tensor_load_to_lds / s_wait_tensorcnt builtin unavailable: cooperative LDS fill fallback"
#endif

#if !__has_builtin(__builtin_amdgcn_permlane16)
#warning "permlane16 builtin unavailable: falling back to __shfl_xor (ds_bpermute)"
#endif

__device__ __forceinline__ v8f wmma_bf16(v16bf a, v16bf b, v8f c) {
    // D = A(16x32 bf16) * B(32x16 bf16) + C(16x16 f32)
    return __builtin_amdgcn_wmma_f32_16x16x32_bf16(
        /*neg_a=*/false, a, /*neg_b=*/false, b,
        /*c_mod=*/(short)0, c, /*reuse_a=*/false, /*reuse_b=*/false);
}

// ---- 16-lane xor-permute reductions (C-matrix rows live in 16-lane halves) --
#if __has_builtin(__builtin_amdgcn_permlane16)
__device__ __forceinline__ float pl16(float v, int sl, int sh) {
    int i = __builtin_bit_cast(int, v);
    i = __builtin_amdgcn_permlane16(i, i, sl, sh, false, false);
    return __builtin_bit_cast(float, i);
}
__device__ __forceinline__ float red_max16(float v) {
    v = fmaxf(v, pl16(v, 0x67452301, 0xEFCDAB89));  // xor 1
    v = fmaxf(v, pl16(v, 0x54761032, 0xDCFE98BA));  // xor 2
    v = fmaxf(v, pl16(v, 0x32107654, 0xBA98FEDC));  // xor 4
    v = fmaxf(v, pl16(v, 0xFEDCBA98, 0x76543210));  // xor 8
    return v;
}
__device__ __forceinline__ float red_sum16(float v) {
    v += pl16(v, 0x67452301, 0xEFCDAB89);
    v += pl16(v, 0x54761032, 0xDCFE98BA);
    v += pl16(v, 0x32107654, 0xBA98FEDC);
    v += pl16(v, 0xFEDCBA98, 0x76543210);
    return v;
}
#else
__device__ __forceinline__ float red_max16(float v) {
    for (int d = 1; d < 16; d <<= 1) v = fmaxf(v, __shfl_xor(v, d));
    return v;
}
__device__ __forceinline__ float red_sum16(float v) {
    for (int d = 1; d < 16; d <<= 1) v += __shfl_xor(v, d);
    return v;
}
#endif

// ---------------------------------------------------------------------------
// fp32 -> bf16 cast, 4 elements/thread
// ---------------------------------------------------------------------------
__global__ __launch_bounds__(256)
void cast_bf16_4(const float* __restrict__ src, __bf16* __restrict__ dst, int n) {
    int i = (blockIdx.x * blockDim.x + threadIdx.x) * 4;
    if (i < n) {
        float4 f = *(const float4*)(src + i);
        v4bf o;
        o[0] = (__bf16)f.x; o[1] = (__bf16)f.y;
        o[2] = (__bf16)f.z; o[3] = (__bf16)f.w;
        *(v4bf*)(dst + i) = o;
    }
}

// ---------------------------------------------------------------------------
// TDM: issue a 2D tile DMA (global -> LDS). Tile = tile_d0 (contig, elems) x
// tile_d1 rows; element = 2 bytes (bf16). Completion tracked via TENSORcnt.
// D# packing per CDNA5 ISA.
// ---------------------------------------------------------------------------
#if USE_TDM
__device__ __forceinline__ void tdm_load_2d(unsigned lds_addr, const __bf16* gptr,
                                            unsigned tensor_d0, unsigned tensor_d1,
                                            unsigned tile_d0, unsigned tile_d1,
                                            unsigned stride0_elems) {
    unsigned long long ga = (unsigned long long)(size_t)gptr;
    v4u g0;
    g0[0] = 1u;                                              // count=1 (valid user D#)
    g0[1] = lds_addr;                                        // LDS byte address
    g0[2] = (unsigned)(ga & 0xffffffffu);                    // global_addr[31:0]
    g0[3] = (unsigned)((ga >> 32) & 0x01ffffffu) | 0x80000000u; // ga[56:32] | type=2
    v8i g1;
    g1[0] = 0x00010000;                                      // data_size=1 (2 bytes)
    g1[1] = (int)(tensor_d0 << 16);                          // tensor_dim0[15:0]
    g1[2] = (int)((tensor_d0 >> 16) | (tensor_d1 << 16));    // d0[31:16] | d1[15:0]
    g1[3] = (int)((tensor_d1 >> 16) | (tile_d0 << 16));      // d1[31:16] | tile_dim0
    g1[4] = (int)(tile_d1 & 0xffffu);                        // tile_dim1, tile_dim2=0
    g1[5] = (int)stride0_elems;                              // tensor_dim0_stride[31:0]
    g1[6] = 0;
    g1[7] = 0;
    v4i z4 = {0, 0, 0, 0};
#if __clang_major__ >= 23
    v8i z8 = {0, 0, 0, 0, 0, 0, 0, 0};
    __builtin_amdgcn_tensor_load_to_lds(g0, g1, z4, z4, z8, 0);
#else
    __builtin_amdgcn_tensor_load_to_lds(g0, g1, z4, z4, 0);
#endif
}
#endif

__device__ __forceinline__ unsigned lds_off(const void* p) {
    // generic LDS pointer: low 32 bits are the LDS byte offset (ISA flat-aperture rule)
    return (unsigned)(unsigned long long)(size_t)p;
}

// ---------------------------------------------------------------------------
// GEMM: C[8192 x 512] = A @ W^T + bias, A/W bf16 row-major, fp32 accum.
// Block tile 128x128, 8 waves (2x4), wave tile 64x32 -> 4x2 wmma accums.
// A/W K-tiles (128x32 bf16) staged in LDS via TDM, double-buffered one K-step
// ahead; fragments read as aligned ds_load_b128.
// STORE_MODE: 0 = bf16 row-major; 1 = bf16 head-transposed Vt[b,h,d,s]; 2 = fp32.
// ---------------------------------------------------------------------------
template <int STORE_MODE>
__global__ __launch_bounds__(256)
void gemm_xwT(const __bf16* __restrict__ A, const __bf16* __restrict__ W,
              const float* __restrict__ bias, void* __restrict__ out) {
    __shared__ __align__(16) __bf16 Asm[2][128 * 32];   // 16 KB
    __shared__ __align__(16) __bf16 Bsm[2][128 * 32];   // 16 KB

    const int tid  = threadIdx.x;
    const int lane = tid & 31;
    const int wid  = tid >> 5;             // 0..7
    const int l15  = lane & 15;
    const int half = lane >> 4;            // 0/1
    const int wm   = wid >> 2;             // 0..1
    const int wn   = wid & 3;              // 0..3
    const int rbase = blockIdx.x * 128 + wm * 64;   // 4 M-tiles of 16
    const int cbase = blockIdx.y * 128 + wn * 32;   // 2 N-tiles of 16
    const __bf16* Ablk = A + (size_t)blockIdx.x * 128 * D_;
    const __bf16* Wblk = W + (size_t)blockIdx.y * 128 * D_;

    constexpr int NK = D_ / 32;            // 16 K-steps

#if USE_TDM
    if (wid == 0) {                        // prologue: K-step 0 into buffer 0
        tdm_load_2d(lds_off(&Asm[0][0]), Ablk, D_, B_ * S_, 32, 128, D_);
        tdm_load_2d(lds_off(&Bsm[0][0]), Wblk, D_, D_,     32, 128, D_);
    }
#else
    {   // cooperative fill of buffer 0: 256 threads x 16 bf16
        const int row = tid >> 1, col = (tid & 1) * 16;
        *(v8bf*)(&Asm[0][tid * 16])     = *(const v8bf*)(Ablk + (size_t)row * D_ + col);
        *(v8bf*)(&Asm[0][tid * 16 + 8]) = *(const v8bf*)(Ablk + (size_t)row * D_ + col + 8);
        *(v8bf*)(&Bsm[0][tid * 16])     = *(const v8bf*)(Wblk + (size_t)row * D_ + col);
        *(v8bf*)(&Bsm[0][tid * 16 + 8]) = *(const v8bf*)(Wblk + (size_t)row * D_ + col + 8);
    }
#endif

    v8f acc[4][2] = {};

    for (int kc = 0; kc < NK; ++kc) {
        const int cur = kc & 1, nxt = cur ^ 1;
        __syncthreads();   // everyone done reading buf[nxt] (from last iter)
#if USE_TDM
        if (wid == 0) {
            if (kc + 1 < NK) {
                const int k1 = (kc + 1) * 32;
                tdm_load_2d(lds_off(&Asm[nxt][0]), Ablk + k1, D_, B_ * S_, 32, 128, D_);
                tdm_load_2d(lds_off(&Bsm[nxt][0]), Wblk + k1, D_, D_,     32, 128, D_);
                __builtin_amdgcn_s_wait_tensorcnt(2);  // cur's 2 DMAs (in-order) done
            } else {
                __builtin_amdgcn_s_wait_tensorcnt(0);
            }
        }
#else
        if (kc + 1 < NK) {
            const int k1 = (kc + 1) * 32;
            const int row = tid >> 1, col = (tid & 1) * 16;
            *(v8bf*)(&Asm[nxt][tid * 16])     = *(const v8bf*)(Ablk + (size_t)row * D_ + k1 + col);
            *(v8bf*)(&Asm[nxt][tid * 16 + 8]) = *(const v8bf*)(Ablk + (size_t)row * D_ + k1 + col + 8);
            *(v8bf*)(&Bsm[nxt][tid * 16])     = *(const v8bf*)(Wblk + (size_t)row * D_ + k1 + col);
            *(v8bf*)(&Bsm[nxt][tid * 16 + 8]) = *(const v8bf*)(Wblk + (size_t)row * D_ + k1 + col + 8);
        }
#endif
        __syncthreads();   // buf[cur] visible to all waves

        const __bf16* As = &Asm[cur][0];
        const __bf16* Bs = &Bsm[cur][0];
        V16 afr[4];
#pragma unroll
        for (int i = 0; i < 4; ++i) {
            const __bf16* ap = As + (wm * 64 + i * 16 + l15) * 32 + 8 * half;
            afr[i].h[0] = *(const v8bf*)(ap);        // K = {0..7}|{8..15}
            afr[i].h[1] = *(const v8bf*)(ap + 16);   // K = 16+{0..7}|{8..15}
        }
        V16 bfr[2];
#pragma unroll
        for (int j = 0; j < 2; ++j) {
            const __bf16* bp = Bs + (wn * 32 + j * 16 + l15) * 32 + 8 * half;
            bfr[j].h[0] = *(const v8bf*)(bp);
            bfr[j].h[1] = *(const v8bf*)(bp + 16);
        }
#pragma unroll
        for (int i = 0; i < 4; ++i)
#pragma unroll
            for (int j = 0; j < 2; ++j)   // 8 independent accums: no WMMA RAW bubbles
                acc[i][j] = wmma_bf16(afr[i].v, bfr[j].v, acc[i][j]);
    }

#pragma unroll
    for (int j = 0; j < 2; ++j) {
        const int col = cbase + j * 16 + l15;
        const float bv = bias[col];
#pragma unroll
        for (int i = 0; i < 4; ++i) {
            const int row0 = rbase + i * 16 + 8 * half;
#pragma unroll
            for (int r = 0; r < 8; ++r) {
                const int row = row0 + r;
                const float v = acc[i][j][r] + bv;
                if (STORE_MODE == 0) {
                    ((__bf16*)out)[(size_t)row * D_ + col] = (__bf16)v;
                } else if (STORE_MODE == 1) {
                    const int b = row >> 11, s = row & (S_ - 1);
                    const int h = col >> 6,  d = col & (DK_ - 1);
                    ((__bf16*)out)[((size_t)(b * H_ + h) * DK_ + d) * S_ + s] = (__bf16)v;
                } else {
                    ((float*)out)[(size_t)row * D_ + col] = v;
                }
            }
        }
    }
}

// ---------------------------------------------------------------------------
// Flash attention (causal). One wave = 16 query rows; key chunks of 64.
// Qp/Kp: bf16 [B,S,D] (head h at col h*64). Vt: bf16 [B,H,DK,S]. fp32 softmax.
// ---------------------------------------------------------------------------
__global__ __launch_bounds__(256)
void flash_attn(const __bf16* __restrict__ Qp, const __bf16* __restrict__ Kp,
                const __bf16* __restrict__ Vt, __bf16* __restrict__ ctx) {
    __shared__ __align__(16) __bf16 pbuf[8][16][64];   // wave-private P tiles

    const int lane = threadIdx.x & 31;
    const int wid  = threadIdx.x >> 5;
    const int l15  = lane & 15;
    const int half = lane >> 4;
    const int bh   = blockIdx.y;               // 0..31
    const int b    = bh >> 3, h = bh & 7;
    const int q0   = blockIdx.x * 128 + wid * 16;

    // Q A-fragments for this wave's 16 rows (K over DK=64 -> 2 fragments)
    const __bf16* qrow = Qp + (size_t)(b * S_ + q0 + l15) * D_ + h * DK_;
    V16 qa[2];
    qa[0].h[0] = *(const v8bf*)(qrow + 8 * half);
    qa[0].h[1] = *(const v8bf*)(qrow + 16 + 8 * half);
    qa[1].h[0] = *(const v8bf*)(qrow + 32 + 8 * half);
    qa[1].h[1] = *(const v8bf*)(qrow + 48 + 8 * half);

    float mrow[8], lrow[8];
#pragma unroll
    for (int r = 0; r < 8; ++r) { mrow[r] = -3.0e38f; lrow[r] = 0.f; }
    v8f cacc[4] = {};

    const int kkmax = (q0 + 15) >> 6;          // last causal key chunk
    for (int kc = 0; kc <= kkmax; ++kc) {
        const int kb = kc * 64;

        // ---- scores: S = Q @ K^T; load all 8 B-frags, then 2 hazard-free sweeps
        V16 kb0[4], kb1[4];
#pragma unroll
        for (int j = 0; j < 4; ++j) {
            const __bf16* krow = Kp + (size_t)(b * S_ + kb + j * 16 + l15) * D_ + h * DK_;
            if (kc < kkmax) __builtin_prefetch(krow + (size_t)64 * D_, 0, 3);
            kb0[j].h[0] = *(const v8bf*)(krow + 8 * half);
            kb0[j].h[1] = *(const v8bf*)(krow + 16 + 8 * half);
            kb1[j].h[0] = *(const v8bf*)(krow + 32 + 8 * half);
            kb1[j].h[1] = *(const v8bf*)(krow + 48 + 8 * half);
        }
        v8f sacc[4] = {};
#pragma unroll
        for (int j = 0; j < 4; ++j) sacc[j] = wmma_bf16(qa[0].v, kb0[j].v, sacc[j]);
#pragma unroll
        for (int j = 0; j < 4; ++j) sacc[j] = wmma_bf16(qa[1].v, kb1[j].v, sacc[j]);

        // ---- online softmax (rows live across 16 lanes of one half) ----
#pragma unroll
        for (int r = 0; r < 8; ++r) {
            const int row = q0 + r + 8 * half;
            float sj[4], mx = -3.0e38f;
#pragma unroll
            for (int j = 0; j < 4; ++j) {
                const int col = kb + j * 16 + l15;
                float s = sacc[j][r] * 0.125f;          // 1/sqrt(DK)
                if (col > row) s = -3.0e38f;            // causal mask
                sj[j] = s;
                mx = fmaxf(mx, s);
            }
            mx = red_max16(mx);
            const float mnew = fmaxf(mrow[r], mx);
            const float corr = __expf(mrow[r] - mnew);
            float ps = 0.f;
#pragma unroll
            for (int j = 0; j < 4; ++j) {
                const float p = __expf(sj[j] - mnew);
                ps += p;
                pbuf[wid][r + 8 * half][j * 16 + l15] = (__bf16)p;
            }
            ps = red_sum16(ps);
            lrow[r] = lrow[r] * corr + ps;
            mrow[r] = mnew;
#pragma unroll
            for (int j2 = 0; j2 < 4; ++j2) cacc[j2][r] *= corr;
        }

        // wave-local LDS RAW fence (no block barrier: waves diverge in trips)
        asm volatile("s_wait_dscnt 0" ::: "memory");

        // ---- ctx += P @ V_chunk (load all B-frags, then 2 hazard-free sweeps)
        const __bf16* prow = &pbuf[wid][l15][0];
        V16 pa[2];
        pa[0].h[0] = *(const v8bf*)(prow + 8 * half);
        pa[0].h[1] = *(const v8bf*)(prow + 16 + 8 * half);
        pa[1].h[0] = *(const v8bf*)(prow + 32 + 8 * half);
        pa[1].h[1] = *(const v8bf*)(prow + 48 + 8 * half);
        V16 vb0[4], vb1[4];
#pragma unroll
        for (int j2 = 0; j2 < 4; ++j2) {
            const __bf16* vrow = Vt + (size_t)(bh * DK_ + j2 * 16 + l15) * S_ + kb;
            if (kc < kkmax) __builtin_prefetch(vrow + 64, 0, 3);
            vb0[j2].h[0] = *(const v8bf*)(vrow + 8 * half);
            vb0[j2].h[1] = *(const v8bf*)(vrow + 16 + 8 * half);
            vb1[j2].h[0] = *(const v8bf*)(vrow + 32 + 8 * half);
            vb1[j2].h[1] = *(const v8bf*)(vrow + 48 + 8 * half);
        }
#pragma unroll
        for (int j2 = 0; j2 < 4; ++j2) cacc[j2] = wmma_bf16(pa[0].v, vb0[j2].v, cacc[j2]);
#pragma unroll
        for (int j2 = 0; j2 < 4; ++j2) cacc[j2] = wmma_bf16(pa[1].v, vb1[j2].v, cacc[j2]);
    }

    // ---- normalize and store context (bf16, [B,S,D]) ----
#pragma unroll
    for (int r = 0; r < 8; ++r) {
        const int row = q0 + r + 8 * half;
        const float inv = 1.0f / lrow[r];
        __bf16* orow = ctx + (size_t)(b * S_ + row) * D_ + h * DK_;
#pragma unroll
        for (int j2 = 0; j2 < 4; ++j2)
            orow[j2 * 16 + l15] = (__bf16)(cacc[j2][r] * inv);
    }
}

// ---------------------------------------------------------------------------
extern "C" void kernel_launch(void* const* d_in, const int* in_sizes, int n_in,
                              void* d_out, int out_size, void* d_ws, size_t ws_size,
                              hipStream_t stream) {
    (void)in_sizes; (void)n_in; (void)out_size; (void)ws_size;
    const float* q   = (const float*)d_in[0];
    const float* k   = (const float*)d_in[1];
    const float* v   = (const float*)d_in[2];
    /* d_in[3] = causal mask; causality is encoded directly in flash_attn */
    const float* w_q = (const float*)d_in[4];
    const float* b_q = (const float*)d_in[5];
    const float* w_k = (const float*)d_in[6];
    const float* b_k = (const float*)d_in[7];
    const float* w_v = (const float*)d_in[8];
    const float* b_v = (const float*)d_in[9];
    const float* w_o = (const float*)d_in[10];
    const float* b_o = (const float*)d_in[11];

    const size_t WB = (size_t)D_ * D_ * sizeof(__bf16);          // 512 KB
    const size_t XB = (size_t)B_ * S_ * D_ * sizeof(__bf16);     // 8 MB
    char* p = (char*)d_ws;                                       // ~58 MB total
    __bf16* wq_b = (__bf16*)(p + 0 * WB);
    __bf16* wk_b = (__bf16*)(p + 1 * WB);
    __bf16* wv_b = (__bf16*)(p + 2 * WB);
    __bf16* wo_b = (__bf16*)(p + 3 * WB);
    char* x = p + 4 * WB;
    __bf16* q_b  = (__bf16*)(x + 0 * XB);
    __bf16* k_b  = (__bf16*)(x + 1 * XB);
    __bf16* v_b  = (__bf16*)(x + 2 * XB);
    __bf16* Qp   = (__bf16*)(x + 3 * XB);
    __bf16* Kp   = (__bf16*)(x + 4 * XB);
    __bf16* Vt   = (__bf16*)(x + 5 * XB);
    __bf16* ctxb = (__bf16*)(x + 6 * XB);

    // casts
    {
        const int nw = D_ * D_;
        dim3 gw((nw / 4 + 255) / 256);
        cast_bf16_4<<<gw, 256, 0, stream>>>(w_q, wq_b, nw);
        cast_bf16_4<<<gw, 256, 0, stream>>>(w_k, wk_b, nw);
        cast_bf16_4<<<gw, 256, 0, stream>>>(w_v, wv_b, nw);
        cast_bf16_4<<<gw, 256, 0, stream>>>(w_o, wo_b, nw);
        const int nx = B_ * S_ * D_;
        dim3 gx((nx / 4 + 255) / 256);
        cast_bf16_4<<<gx, 256, 0, stream>>>(q, q_b, nx);
        cast_bf16_4<<<gx, 256, 0, stream>>>(k, k_b, nx);
        cast_bf16_4<<<gx, 256, 0, stream>>>(v, v_b, nx);
    }

    // projections: M = B*S = 8192, N = K = 512
    dim3 gg(B_ * S_ / 128, D_ / 128);   // (64, 4)
    gemm_xwT<0><<<gg, 256, 0, stream>>>(q_b, wq_b, b_q, Qp);
    gemm_xwT<0><<<gg, 256, 0, stream>>>(k_b, wk_b, b_k, Kp);
    gemm_xwT<1><<<gg, 256, 0, stream>>>(v_b, wv_b, b_v, Vt);   // head-transposed

    // causal flash attention
    flash_attn<<<dim3(S_ / 128, B_ * H_), 256, 0, stream>>>(Qp, Kp, Vt, ctxb);

    // output projection -> fp32 d_out
    gemm_xwT<2><<<gg, 256, 0, stream>>>(ctxb, wo_b, b_o, d_out);
}
